// MInferModel_30253749633437
// MI455X (gfx1250) — matrix operand order repaired and checked by reading, not compiled, not run on hardware
//
#include <hip/hip_runtime.h>

typedef __attribute__((ext_vector_type(16))) _Float16 v16h;
typedef __attribute__((ext_vector_type(8)))  _Float16 h8;
typedef __attribute__((ext_vector_type(8)))  float    v8f;

#define DEVFN __device__ __forceinline__

#if __has_builtin(__builtin_amdgcn_sched_barrier)
#define SCHED_FENCE() __builtin_amdgcn_sched_barrier(0)
#else
#define SCHED_FENCE()
#endif

constexpr int H = 32;
constexpr int S = 2048;
constexpr int D = 128;
constexpr int LAST_Q = 64;
constexpr int V_TOPK = 256;
constexpr int S_TOPK = 512;
constexpr int N_SINK = 4;
constexpr int N_LOCAL = 64;
constexpr float SCALE = 0.08838834764831845f;   // 1/sqrt(128)
constexpr int NW = S / 32;                      // mask words per head

DEVFN v8f wmma16(v16h a, v16h b, v8f c) {
  return __builtin_amdgcn_wmma_f32_16x16x32_f16(false, a, false, b, (short)0, c,
                                                false, false);
}

// Build f16 A-operand row chunk: this lane's row, d-offsets [0..7] and [16..23]
// relative to rowp (caller adds +8 for the high lane-half).
DEVFN v16h load_a_f16(const _Float16* rowp) {
  h8 lo = *(const h8*)(rowp);
  h8 hi = *(const h8*)(rowp + 16);
  v16h a;
#pragma unroll
  for (int i = 0; i < 8; ++i) { a[i] = lo[i]; a[8 + i] = hi[i]; }
  return a;
}

// ---------------------------------------------------------------------------
// Preprocess: K,V -> f16 row-major; Q -> f16 transposed per head [h][d][s]
// ---------------------------------------------------------------------------
__global__ __launch_bounds__(256) void cvt_kv(const float* __restrict__ k,
                                              const float* __restrict__ v,
                                              _Float16* __restrict__ kf,
                                              _Float16* __restrict__ vf, int n) {
  int i = blockIdx.x * 256 + threadIdx.x;
  if (i < n) { kf[i] = (_Float16)k[i]; vf[i] = (_Float16)v[i]; }
}

__global__ __launch_bounds__(256) void transpose_q(const float* __restrict__ q,
                                                   _Float16* __restrict__ qT, int n) {
  int i = blockIdx.x * 256 + threadIdx.x;
  if (i < n) {
    int d = i & (D - 1);
    int s = (i >> 7) & (S - 1);
    int h = i >> 18;
    qT[((size_t)h * D + d) * S + s] = (_Float16)q[i];
  }
}

// ---------------------------------------------------------------------------
// Stage 1: estimate vertical/slash importance from last 64 queries.
// One block (4 waves) per head; each wave owns 16 query rows.
// Sᵀ-formulation: C layout = (VGPR=key, lane=q).
// ---------------------------------------------------------------------------
__global__ __launch_bounds__(128) void estimate(const _Float16* __restrict__ kf,
                                                const _Float16* __restrict__ qT,
                                                float* __restrict__ vertical,
                                                float* __restrict__ slash) {
  __shared__ float vert[S];
  __shared__ float sl[S];
  const int h = blockIdx.x;
  const int tid = threadIdx.x;
  const int lane = tid & 31;
  const int wave = tid >> 5;
  for (int i = tid; i < S; i += 128) { vert[i] = 0.f; sl[i] = 0.f; }
  __syncthreads();

  const _Float16* kh = kf + (size_t)h * S * D;
  const _Float16* qh = qT + (size_t)h * D * S;
  const int q0 = S - LAST_Q + wave * 16;

  v16h Qb[4];
#pragma unroll
  for (int dc = 0; dc < 4; ++dc)
    for (int s = 0; s < 16; ++s)
      Qb[dc][s] = qh[(size_t)(dc * 32 + (s & 7) + ((s & 8) << 1) + ((lane & 16) >> 1)) * S + q0 + (lane & 15)];

  const int qme = q0 + (lane & 15);
  const int lhalf = (lane & 16) ? 8 : 0;
  const int cend = q0 + 16;
  v8f zero = {};

  auto qk_tiles = [&](int c0, v8f& s0, v8f& s1) {
    const _Float16* krow = kh + (size_t)(c0 + (lane & 15)) * D + lhalf;
    v16h a[8];                       // cluster all 16 b128 loads
#pragma unroll
    for (int dc = 0; dc < 4; ++dc) {
      a[dc] = load_a_f16(krow + dc * 32);
      a[4 + dc] = load_a_f16(krow + 16 * D + dc * 32);
    }
    SCHED_FENCE();                   // keep loads issued before the WMMA chain
    s0 = zero; s1 = zero;
#pragma unroll
    for (int dc = 0; dc < 4; ++dc) {
      s0 = wmma16(a[dc], Qb[dc], s0);
      s1 = wmma16(a[4 + dc], Qb[dc], s1);
    }
  };

  // pass 1: row max / row sum (online)
  float mrun = -3.0e38f, lrun = 0.f;
  for (int c0 = 0; c0 < cend; c0 += 32) {
    v8f s0, s1;
    qk_tiles(c0, s0, s1);
    const int kbase = c0 + lhalf;
    float x0[8], x1[8];
#pragma unroll
    for (int j = 0; j < 8; ++j) {
      x0[j] = (qme >= kbase + j)      ? s0[j] * SCALE : -1.0e30f;
      x1[j] = (qme >= kbase + 16 + j) ? s1[j] * SCALE : -1.0e30f;
    }
    float tm = -1.0e30f;
#pragma unroll
    for (int j = 0; j < 8; ++j) { tm = fmaxf(tm, x0[j]); tm = fmaxf(tm, x1[j]); }
    tm = fmaxf(tm, __shfl_xor(tm, 16, 32));
    float mnew = fmaxf(mrun, tm);
    float corr = __expf(mrun - mnew);
    float psum = 0.f;
#pragma unroll
    for (int j = 0; j < 8; ++j) {
      psum += (x0[j] <= -1.0e29f) ? 0.f : __expf(x0[j] - mnew);
      psum += (x1[j] <= -1.0e29f) ? 0.f : __expf(x1[j] - mnew);
    }
    psum += __shfl_xor(psum, 16, 32);
    lrun = lrun * corr + psum;
    mrun = mnew;
  }
  const float mfin = mrun;
  const float linv = 1.0f / lrun;

  // pass 2: p = exp(s-m)/l, accumulate vertical[key] and slash[q-key]
  for (int c0 = 0; c0 < cend; c0 += 32) {
    v8f s0, s1;
    qk_tiles(c0, s0, s1);
    const int kbase = c0 + lhalf;
    float p0[8], p1[8];
#pragma unroll
    for (int j = 0; j < 8; ++j) {
      p0[j] = (qme >= kbase + j)      ? __expf(s0[j] * SCALE - mfin) * linv : 0.f;
      p1[j] = (qme >= kbase + 16 + j) ? __expf(s1[j] * SCALE - mfin) * linv : 0.f;
    }
#pragma unroll
    for (int j = 0; j < 8; ++j) {
      float r0 = p0[j], r1 = p1[j];
#pragma unroll
      for (int m = 1; m < 16; m <<= 1) {
        r0 += __shfl_xor(r0, m, 32);
        r1 += __shfl_xor(r1, m, 32);
      }
      if ((lane & 15) == 0) {
        atomicAdd(&vert[kbase + j], r0);
        atomicAdd(&vert[kbase + 16 + j], r1);
      }
      int d0 = qme - (kbase + j);
      int d1 = qme - (kbase + 16 + j);
      if (d0 >= 0 && p0[j] > 0.f) atomicAdd(&sl[d0], p0[j]);
      if (d1 >= 0 && p1[j] > 0.f) atomicAdd(&sl[d1], p1[j]);
    }
  }
  __syncthreads();
  for (int i = tid; i < S; i += 128) {
    vertical[(size_t)h * S + i] = vert[i];
    slash[(size_t)h * S + i] = sl[i];
  }
}

// ---------------------------------------------------------------------------
// Stage 2: exact top-k -> bitmasks via 32-step radix select; ties broken by
// lowest index (matches jax.lax.top_k). blockIdx.y: 0=vertical, 1=slash.
// ---------------------------------------------------------------------------
__global__ __launch_bounds__(256) void topk(const float* __restrict__ vertical,
                                            const float* __restrict__ slash,
                                            unsigned* __restrict__ vmaskg,
                                            unsigned* __restrict__ smaskg) {
  const int h = blockIdx.x;
  const int type = blockIdx.y;
  const float* src = (type == 0) ? (vertical + (size_t)h * S) : (slash + (size_t)h * S);
  unsigned* dst = (type == 0) ? (vmaskg + h * NW) : (smaskg + h * NW);
  const int K = (type == 0) ? V_TOPK : S_TOPK;
  const int npre = (type == 0) ? N_SINK : N_LOCAL;

  __shared__ unsigned u[S];
  __shared__ int cnt;
  const int tid = threadIdx.x;
  for (int i = tid; i < S; i += 256) {
    float f = (i < npre) ? __builtin_inff() : src[i];
    unsigned b = __float_as_uint(f);
    u[i] = b ^ ((b & 0x80000000u) ? 0xFFFFFFFFu : 0x80000000u);
  }
  unsigned T = 0u;
  for (int bit = 31; bit >= 0; --bit) {
    __syncthreads();
    if (tid == 0) cnt = 0;
    __syncthreads();
    unsigned cand = T | (1u << bit);
    int local = 0;
    for (int i = tid; i < S; i += 256) local += (u[i] >= cand) ? 1 : 0;
    if (local) atomicAdd(&cnt, local);
    __syncthreads();
    if (cnt >= K) T = cand;
  }
  __syncthreads();
  if (tid == 0) cnt = 0;
  __syncthreads();
  int local = 0;
  for (int i = tid; i < S; i += 256) local += (u[i] > T) ? 1 : 0;
  if (local) atomicAdd(&cnt, local);
  __syncthreads();
  if (tid == 0) {
    int rem = K - cnt;
    for (int wi = 0; wi < NW; ++wi) {
      unsigned word = 0u;
      for (int bj = 0; bj < 32; ++bj) {
        unsigned uv = u[wi * 32 + bj];
        bool sel = (uv > T);
        if (!sel && uv == T && rem > 0) { sel = true; --rem; }
        if (sel) word |= (1u << bj);
      }
      dst[wi] = word;
    }
  }
}

// ---------------------------------------------------------------------------
// Stage 3: masked flash attention with tile skipping + software pipelining.
// Block = 4 waves, 64 query rows; wave = 16 query rows; key tiles of 32.
// Per tile: issue V loads + next-tile K loads FIRST (sched-fenced), then the
// QK WMMA chain (kA already in flight), then softmax, then PV.  In-order
// LOADcnt retirement means the PV wait for V leaves the K prefetch pending.
// ---------------------------------------------------------------------------
__global__ __launch_bounds__(128) void sparse_attn(const _Float16* __restrict__ kf,
                                                   const _Float16* __restrict__ vf,
                                                   const _Float16* __restrict__ qT,
                                                   const unsigned* __restrict__ vmaskg,
                                                   const unsigned* __restrict__ smaskg,
                                                   float* __restrict__ out) {
  __shared__ unsigned vm[NW];
  __shared__ unsigned sm[NW];
  const int h = blockIdx.y;
  const int tid = threadIdx.x;
  const int lane = tid & 31;
  const int wave = tid >> 5;
  const int q0 = blockIdx.x * 64 + wave * 16;

  if (tid < NW) vm[tid] = vmaskg[h * NW + tid];
  else if (tid < 2 * NW) sm[tid - NW] = smaskg[h * NW + tid - NW];
  __syncthreads();

  const _Float16* kh = kf + (size_t)h * S * D;
  const _Float16* vh = vf + (size_t)h * S * D;
  const _Float16* qh = qT + (size_t)h * D * S;

  v16h Qb[4];                                   // resident Qᵀ B-operands
#pragma unroll
  for (int dc = 0; dc < 4; ++dc)
    for (int s = 0; s < 16; ++s)
      Qb[dc][s] = qh[(size_t)(dc * 32 + (s & 7) + ((s & 8) << 1) + ((lane & 16) >> 1)) * S + q0 + (lane & 15)];

  v8f zero = {};
  v8f acc[8];
#pragma unroll
  for (int t = 0; t < 8; ++t) acc[t] = zero;

  float mrun = -3.0e38f, lrun = 0.f;
  const int qme = q0 + (lane & 15);
  const int lhalf = (lane & 16) ? 8 : 0;
  const int ntiles = (q0 + 16 + 31) >> 5;       // causal tile count for this wave

  // ---- live-tile bitmask (wave-uniform) ----
  unsigned long long lm = 0ull;
  for (int ti = 0; ti < ntiles; ++ti) {
    const int c0 = ti * 32;
    bool vhit = vm[ti] != 0u;
    int dhi = q0 + 15 - c0;
    int dlo = q0 - c0 - 31; if (dlo < 0) dlo = 0;
    bool shit = false;
    for (int w = dlo >> 5; w <= (dhi >> 5); ++w) {
      unsigned lo = (w == (dlo >> 5)) ? (unsigned)(dlo & 31) : 0u;
      unsigned hi = (w == (dhi >> 5)) ? (unsigned)(dhi & 31) : 31u;
      unsigned msk = ((hi == 31u) ? 0xFFFFFFFFu : ((1u << (hi + 1u)) - 1u)) &
                     ~((1u << lo) - 1u);
      if (sm[w] & msk) shit = true;
    }
    if (vhit || shit) lm |= (1ull << ti);
  }

  auto loadK = [&](v16h (&dst)[8], int c0) {
    const _Float16* krow = kh + (size_t)(c0 + (lane & 15)) * D + lhalf;
#pragma unroll
    for (int dc = 0; dc < 4; ++dc) {
      dst[dc] = load_a_f16(krow + dc * 32);
      dst[4 + dc] = load_a_f16(krow + 16 * D + dc * 32);
    }
  };

  // process one live tile; prefetch next tile's K into kB if cn >= 0
  auto process = [&](int c0, const v16h (&kA)[8], int cn, v16h (&kB)[8]) {
    // Issue ALL long-latency loads up front: this tile's V, then next K.
    v16h vb[8];
    const _Float16* vrow = vh + (size_t)c0 * D + (lane & 15);   // B lane = d column, elements = key
#pragma unroll
    for (int t = 0; t < 8; ++t)
      for (int s = 0; s < 16; ++s)
        vb[t][s] = vrow[(size_t)((s & 7) + ((s & 8) << 1) + ((lane & 16) >> 1)) * D + t * 16];
    if (cn >= 0) loadK(kB, cn);
    SCHED_FENCE();                   // do not sink these loads to their uses

    // Sᵀ = K_tile · Qᵀ : two 16(key)x16(q) tiles, K-dim = 128 (4 chunks)
    v8f s0 = zero, s1 = zero;
#pragma unroll
    for (int dc = 0; dc < 4; ++dc) {
      s0 = wmma16(kA[dc], Qb[dc], s0);
      s1 = wmma16(kA[4 + dc], Qb[dc], s1);
    }

    const int kbase = c0 + lhalf;
    float x0[8], x1[8];
#pragma unroll
    for (int j = 0; j < 8; ++j) {
      int k0 = kbase + j, k1 = k0 + 16;
      int d0 = qme - k0, d1 = qme - k1;
      int dc0 = d0 < 0 ? 0 : d0, dc1 = d1 < 0 ? 0 : d1;
      bool keep0 = (d0 >= 0) &&
          ((((vm[k0 >> 5] >> (k0 & 31)) | (sm[dc0 >> 5] >> (dc0 & 31))) & 1u) != 0u);
      bool keep1 = (d1 >= 0) &&
          ((((vm[k1 >> 5] >> (k1 & 31)) | (sm[dc1 >> 5] >> (dc1 & 31))) & 1u) != 0u);
      x0[j] = keep0 ? s0[j] * SCALE : -1.0e30f;
      x1[j] = keep1 ? s1[j] * SCALE : -1.0e30f;
    }

    // online softmax (state replicated in both lane-halves)
    float tm = -1.0e30f;
#pragma unroll
    for (int j = 0; j < 8; ++j) { tm = fmaxf(tm, x0[j]); tm = fmaxf(tm, x1[j]); }
    tm = fmaxf(tm, __shfl_xor(tm, 16, 32));
    float mnew = fmaxf(mrun, tm);
    float corr = __expf(mrun - mnew);
    float p0[8], p1[8], psum = 0.f;
#pragma unroll
    for (int j = 0; j < 8; ++j) {
      p0[j] = (x0[j] <= -1.0e29f) ? 0.f : __expf(x0[j] - mnew);
      p1[j] = (x1[j] <= -1.0e29f) ? 0.f : __expf(x1[j] - mnew);
      psum += p0[j] + p1[j];
    }
    psum += __shfl_xor(psum, 16, 32);
    lrun = lrun * corr + psum;
    mrun = mnew;

    // rescale accumulators: broadcast corr from lane=q layout to VGPR=q layout
    float cb[8];
#pragma unroll
    for (int i = 0; i < 8; ++i)
      cb[i] = __shfl(corr, (lane & 16) ? (i + 8) : i, 32);
#pragma unroll
    for (int t = 0; t < 8; ++t)
#pragma unroll
      for (int i = 0; i < 8; ++i) acc[t][i] *= cb[i];

    // P: per-lane f16 pack of the two Sᵀ tiles == PV A-operand layout
    v16h pa;
#pragma unroll
    for (int j = 0; j < 8; ++j) {
      pa[j] = (_Float16)p0[j];
      pa[8 + j] = (_Float16)p1[j];
    }
#pragma unroll
    for (int t = 0; t < 8; ++t) acc[t] = wmma16(pa, vb[t], acc[t]);
  };

  // ---- ping-pong pipelined loop over live tiles ----
  v16h ka[8], kb[8];
  int t0 = lm ? (__ffsll(lm) - 1) : -1;
  if (t0 >= 0) loadK(ka, t0 * 32);
  while (t0 >= 0) {
    lm &= lm - 1;                                // clear t0
    int t1 = lm ? (__ffsll(lm) - 1) : -1;
    process(t0 * 32, ka, (t1 >= 0) ? t1 * 32 : -1, kb);
    if (t1 < 0) break;
    lm &= lm - 1;                                // clear t1
    int t2 = lm ? (__ffsll(lm) - 1) : -1;
    process(t1 * 32, kb, (t2 >= 0) ? t2 * 32 : -1, ka);
    t0 = t2;
  }

  float lb[8];
#pragma unroll
  for (int i = 0; i < 8; ++i)
    lb[i] = 1.0f / __shfl(lrun, (lane & 16) ? (i + 8) : i, 32);

#pragma unroll
  for (int t = 0; t < 8; ++t) {
    int dcol = t * 16 + (lane & 15);
#pragma unroll
    for (int i = 0; i < 8; ++i) {
      int qrow = q0 + i + lhalf;
      out[((size_t)h * S + qrow) * D + dcol] = acc[t][i] * lb[i];
    }
  }
}

// ---------------------------------------------------------------------------
extern "C" void kernel_launch(void* const* d_in, const int* in_sizes, int n_in,
                              void* d_out, int out_size, void* d_ws, size_t ws_size,
                              hipStream_t stream) {
  (void)in_sizes; (void)n_in; (void)out_size; (void)ws_size;
  const float* q = (const float*)d_in[0];
  const float* k = (const float*)d_in[1];
  const float* v = (const float*)d_in[2];
  float* out = (float*)d_out;

  char* ws = (char*)d_ws;
  const size_t nkv = (size_t)H * S * D;                 // 8,388,608
  _Float16* kf = (_Float16*)(ws);
  _Float16* vf = (_Float16*)(ws + nkv * 2);
  _Float16* qT = (_Float16*)(ws + nkv * 4);
  float* vertical = (float*)(ws + nkv * 6);
  float* slash    = (float*)(ws + nkv * 6 + (size_t)H * S * 4);
  unsigned* vmaskg = (unsigned*)(ws + nkv * 6 + (size_t)H * S * 8);
  unsigned* smaskg = (unsigned*)(ws + nkv * 6 + (size_t)H * S * 8 + (size_t)H * NW * 4);

  const int n = (int)nkv;
  cvt_kv<<<(n + 255) / 256, 256, 0, stream>>>(k, v, kf, vf, n);
  transpose_q<<<(n + 255) / 256, 256, 0, stream>>>(q, qT, n);
  estimate<<<H, 128, 0, stream>>>(kf, qT, vertical, slash);
  topk<<<dim3(H, 2), 256, 0, stream>>>(vertical, slash, vmaskg, smaskg);
  sparse_attn<<<dim3(S / 64, H), 128, 0, stream>>>(kf, vf, qT, vmaskg, smaskg, out);
}
